// RSSM_83425444757789
// MI455X (gfx1250) — matrix operand order repaired and checked by reading, not compiled, not run on hardware
//
#include <hip/hip_runtime.h>
#include <hip/hip_bf16.h>
#include <stdint.h>

#define T_    64
#define N_    1024
#define H_    32
#define XC_   10
#define AC_   6
#define ZS_   32
#define ZC_   32
#define ZF_   1024
#define GIN_  1030
#define KENC_ 64   // encoder K padded 42 -> 64

typedef __attribute__((ext_vector_type(16))) _Float16 v16h;
typedef __attribute__((ext_vector_type(8)))  float    v8f;
typedef __attribute__((ext_vector_type(4)))  uint32_t u32x4;
typedef __attribute__((ext_vector_type(8)))  uint32_t u32x8;

// ---- CDNA5 WMMA 16x16x32 f16 fragment index helpers (ISA 7.12.2) ----
// A (16-bit 16x32): lane L holds row M=L&15; element j -> K = ((L<16)?0:16)+j (contiguous; this code's choice)
__device__ __forceinline__ int a_kidx(int lane, int j) {
  return j + ((lane < 16) ? 0 : 16);         // contiguous K, matches B
}
// B (16-bit 32x16): lane L -> N=L&15, element j -> K = ((L<16)?0:16)+j (contiguous; this code's choice)
// C/D (f32 16x16):  lane L -> N=L&15, element r -> M = r + ((L<16)?0:8)

__device__ __forceinline__ uint32_t hashu(uint32_t x) {
  x ^= x >> 16; x *= 0x7feb352du;
  x ^= x >> 15; x *= 0x846ca68bu;
  x ^= x >> 16; return x;
}
__device__ __forceinline__ float gumbel4(uint32_t a, uint32_t b, uint32_t c, uint32_t d) {
  uint32_t h = hashu(a * 0x9E3779B9u + hashu(b * 0x85EBCA6Bu + hashu(c * 0xC2B2AE35u + d)));
  float u = (float)(h >> 8) * (1.0f / 16777216.0f) + 1e-12f;
  return -__logf(-__logf(u));
}

// ---------------------------------------------------------------------------
// TDM: async load of the 16x10 f32 embedder slab x[t, n0:n0+16, :] into LDS.
// 2-group descriptor form (VADDR2/3 = NULL). Issued once per wave (EXEC
// ignored for tensor ops); guarded at wave granularity.
// ---------------------------------------------------------------------------
__device__ __forceinline__ void tdm_load_e(const float* gsrc, uint32_t ldsAddr) {
  uint64_t ga = (uint64_t)(uintptr_t)gsrc;
  u32x4 g0;
  g0[0] = 1u;                                   // count=1, is_restore=0, no gather
  g0[1] = ldsAddr;                              // lds_addr (bytes)
  g0[2] = (uint32_t)ga;                         // global_addr[31:0]
  g0[3] = (uint32_t)((ga >> 32) & 0x01FFFFFFu)  // global_addr[56:32]
          | (2u << 30);                         // type = 2 ("image")
  u32x8 g1;
  g1[0] = 2u << 16;         // workgroup_mask=0, data_size=2 (4 bytes)
  g1[1] = XC_ << 16;        // tensor_dim0[15:0] = 10   (bits 63:48)
  g1[2] = 16u << 16;        // tensor_dim0[31:16]=0 | tensor_dim1[15:0]=16
  g1[3] = XC_ << 16;        // tensor_dim1[31:16]=0 | tile_dim0=10
  g1[4] = 16u;              // tile_dim1=16, tile_dim2=0
  g1[5] = XC_;              // tensor_dim0_stride = 10 (low 32)
  g1[6] = 0u;
  g1[7] = 0u;               // tensor_dim1_stride = 0 (unused, single 2D tile)
  asm volatile("tensor_load_to_lds %0, %1" :: "s"(g0), "s"(g1) : "memory");
}

// ---------------------------------------------------------------------------
// Encoder slice: logits = [h || e_t] @ enc_W^T + enc_b (K=42 padded to 64),
// groups [sBeg,sEnd) handled by this wave; Gumbel-max sample -> one-hot z.
// ---------------------------------------------------------------------------
__device__ void encode_sample(
    int lane, int n0, int t, int sBeg, int sEnd,
    const float* __restrict__ ebuf,       // LDS [16][10] e_t slab
    const _Float16* __restrict__ encWh,   // [1024][64] f16 (padded)
    const float* __restrict__ enc_b,      // [1024]
    const float* __restrict__ hbuf,       // LDS [16][32]
    float* __restrict__ stage,            // LDS [16][32] (per wave)
    unsigned char* __restrict__ idxb,     // LDS [16][32] (shared)
    unsigned char* __restrict__ idxWs,    // global [T*N][32]
    float* __restrict__ zout)             // d_out z region base
{
  const int m     = lane & 15;
  const int kbase = (lane < 16) ? 0 : 16;
  const int rbase = (lane < 16) ? 0 : 8;

  // A fragments: A0 = h (K 0..31), A1 = e_t padded (K 32..63, valid < 42)
  v16h A0, A1;
  for (int j = 0; j < 16; ++j) {
    int K = a_kidx(lane, j);
    A0[j] = (_Float16)hbuf[m * H_ + K];
    int K1 = K + 32;
    float ev = 0.0f;
    if (K1 < (H_ + XC_)) ev = ebuf[m * XC_ + (K1 - H_)];
    A1[j] = (_Float16)ev;
  }

  for (int s = sBeg; s < sEnd; ++s) {
    if (s + 1 < sEnd) __builtin_prefetch(encWh + (size_t)((2 * s + 2) * 16) * KENC_, 0, 1);
    for (int sub = 0; sub < 2; ++sub) {
      int nt  = 2 * s + sub;
      int col = nt * 16 + m;
      const v16h B0 = *reinterpret_cast<const v16h*>(encWh + (size_t)col * KENC_ + kbase);
      const v16h B1 = *reinterpret_cast<const v16h*>(encWh + (size_t)col * KENC_ + 32 + kbase);
      v8f acc = {};
      acc = __builtin_amdgcn_wmma_f32_16x16x32_f16(false, A0, false, B0, (short)0, acc, false, false);
      acc = __builtin_amdgcn_wmma_f32_16x16x32_f16(false, A1, false, B1, (short)0, acc, false, false);
      float eb = enc_b[col];
      for (int r = 0; r < 8; ++r)
        stage[(rbase + r) * ZC_ + sub * 16 + m] = acc[r] + eb;
    }
    // Gumbel-max categorical sample (ST forward value == one-hot)
    if (lane < 16) {
      int row = lane;
      int n   = n0 + row;
      float best = -1e30f; int ci = 0;
      for (int c = 0; c < ZC_; ++c) {
        float v = stage[row * ZC_ + c] + gumbel4((uint32_t)t, (uint32_t)n, (uint32_t)s, (uint32_t)c);
        if (v > best) { best = v; ci = c; }
      }
      idxb[row * ZS_ + s] = (unsigned char)ci;
      idxWs[((size_t)t * N_ + n) * ZS_ + s] = (unsigned char)ci;
      float4* zp = reinterpret_cast<float4*>(zout + ((size_t)t * N_ + n) * ZF_ + s * ZC_);
      for (int q = 0; q < 8; ++q) {
        float4 v4 = make_float4(0.f, 0.f, 0.f, 0.f);
        if ((ci >> 2) == q) ((float*)&v4)[ci & 3] = 1.0f;
        zp[q] = v4;
      }
    }
  }
}

// ---------------------------------------------------------------------------
// Sequential scan: block = 4 waves owning 16 sequences for all 64 steps.
//   wave w: encoder groups [8w, 8w+8), gather rows [4w, 4w+4)
//   wave 0: GRU (6 WMMAs) + gates + h update, TDM e-prefetch pipeline.
// ---------------------------------------------------------------------------
__global__ __launch_bounds__(128) void rssm_scan(
    const float* __restrict__ x, const float* __restrict__ a,
    const float* __restrict__ h0, const float* __restrict__ enc_b,
    const float* __restrict__ gru_Whh, const float* __restrict__ gru_bih,
    const float* __restrict__ gru_bhh,
    const _Float16* __restrict__ encWh, const float* __restrict__ WihT,
    float* __restrict__ hOut, unsigned char* __restrict__ idxWs,
    float* __restrict__ zOut)
{
  __shared__ float         s_h[16 * H_];
  __shared__ float         s_gx[16 * 96];
  __shared__ float         s_stage[4][16 * ZC_];
  __shared__ unsigned char s_idx[16 * ZS_];
  __shared__ float         s_e[2][16 * XC_];

  const int wave = threadIdx.x >> 5;
  const int lane = threadIdx.x & 31;
  const int n0   = blockIdx.x * 16;
  float* stage = s_stage[wave];
  const int m     = lane & 15;
  const int kbase = (lane < 16) ? 0 : 16;
  const int rbase = (lane < 16) ? 0 : 8;

  // Kick TDM prefetch of e[0] (wave 0 only; tensor op issues per-wave)
  if (wave == 0)
    tdm_load_e(x + (size_t)n0 * XC_, (uint32_t)(uintptr_t)&s_e[0][0]);

  // h0 -> LDS + hOut[t=0]
  for (int i = threadIdx.x; i < 16 * H_; i += 128) {
    float v = h0[(size_t)n0 * H_ + i];
    s_h[i] = v;
    hOut[(size_t)n0 * H_ + i] = v;
  }

  // Whh B-fragments: constant across time, register resident
  v16h whhB[6];
  for (int tile = 0; tile < 6; ++tile) {
    int g = tile * 16 + m;
    for (int j = 0; j < 16; ++j)
      whhB[tile][j] = (_Float16)gru_Whh[(size_t)g * H_ + kbase + j];
  }
  float bih0 = gru_bih[lane], bih1 = gru_bih[lane + 32], bih2 = gru_bih[lane + 64];

  if (wave == 0) {
    tdm_load_e(x + ((size_t)N_ + n0) * XC_, (uint32_t)(uintptr_t)&s_e[1][0]);  // e[1]
    __builtin_amdgcn_s_wait_tensorcnt(1);  // e[0] landed, e[1] in flight
  }
  __syncthreads();

  // z0 = enc(h0, e[0])
  encode_sample(lane, n0, 0, wave * 8, wave * 8 + 8, s_e[0], encWh, enc_b,
                s_h, stage, s_idx, idxWs, zOut);

  for (int t = 1; t < T_; ++t) {
    __syncthreads();   // idx(t-1) visible; e-slot (t+1)&1 free for reuse

    // ---- gx = za@Wih^T + bih : one-hot gather + action dot (4 rows/wave)
    for (int r = wave * 4; r < wave * 4 + 4; ++r) {
      float a0 = bih0, a1 = bih1, a2 = bih2;   // lane owns g = lane, +32, +64
      for (int s = 0; s < ZS_; ++s) {
        int col = s * ZC_ + (int)s_idx[r * ZS_ + s];
        const float* wc = WihT + (size_t)col * 96;
        a0 += wc[lane]; a1 += wc[lane + 32]; a2 += wc[lane + 64];
      }
      const float* ap = a + (((size_t)(t - 1) * N_) + (n0 + r)) * AC_;
      for (int c = 0; c < AC_; ++c) {
        float av = ap[c];
        const float* wc = WihT + (size_t)(ZF_ + c) * 96;
        a0 += av * wc[lane]; a1 += av * wc[lane + 32]; a2 += av * wc[lane + 64];
      }
      s_gx[r * 96 + lane]      = a0;
      s_gx[r * 96 + lane + 32] = a1;
      s_gx[r * 96 + lane + 64] = a2;
    }
    __syncthreads();

    if (wave == 0) {
      // ---- gh = h_prev @ Whh^T via WMMA (6 tiles of 16 gate columns)
      v16h Ah;
      for (int j = 0; j < 16; ++j)
        Ah[j] = (_Float16)s_h[m * H_ + a_kidx(lane, j)];
      v8f gh[6];
      for (int tile = 0; tile < 6; ++tile) {
        v8f z8 = {};
        gh[tile] = __builtin_amdgcn_wmma_f32_16x16x32_f16(false, Ah, false, whhB[tile],
                                                          (short)0, z8, false, false);
      }
      // ---- fused gates + h update in D-fragment layout
      for (int tr = 0; tr < 2; ++tr) {
        int k = tr * 16 + m;
        float br = gru_bhh[k], bu = gru_bhh[32 + k], bn = gru_bhh[64 + k];
        for (int e8 = 0; e8 < 8; ++e8) {
          int row = rbase + e8;
          float xr = s_gx[row * 96 + k];
          float xz = s_gx[row * 96 + 32 + k];
          float xn = s_gx[row * 96 + 64 + k];
          float hr = gh[tr][e8] + br;
          float hz = gh[tr + 2][e8] + bu;
          float hn = gh[tr + 4][e8] + bn;
          float rg = 1.0f / (1.0f + __expf(-(xr + hr)));
          float ug = 1.0f / (1.0f + __expf(-(xz + hz)));
          float ng = tanhf(xn + rg * hn);
          float hp = s_h[row * H_ + k];
          float hv = (1.0f - ug) * ng + ug * hp;
          s_h[row * H_ + k] = hv;
          hOut[((size_t)t * N_ + n0 + row) * H_ + k] = hv;
        }
      }
      // ---- TDM pipeline: fetch e[t+1] while e[t] must be complete
      if (t < T_ - 1) {
        tdm_load_e(x + ((size_t)(t + 1) * N_ + n0) * XC_,
                   (uint32_t)(uintptr_t)&s_e[(t + 1) & 1][0]);
        __builtin_amdgcn_s_wait_tensorcnt(1);   // e[t] done, e[t+1] in flight
      } else {
        __builtin_amdgcn_s_wait_tensorcnt(0);   // drain
      }
    }
    __syncthreads();   // h_t and e[t] visible to all waves

    // ---- z_t = enc(h_t, e_t)
    encode_sample(lane, n0, t, wave * 8, wave * 8 + 8, s_e[t & 1], encWh, enc_b,
                  s_h, stage, s_idx, idxWs, zOut);
  }
}

// ---------------------------------------------------------------------------
// Post-scan heads: dynamics (WMMA) + z_post sample, decoder/reward/continue.
// ---------------------------------------------------------------------------
__global__ __launch_bounds__(256) void rssm_heads(
    const float* __restrict__ hOut, const unsigned char* __restrict__ idxWs,
    const _Float16* __restrict__ dynWh, const float* __restrict__ dyn_b,
    const float* __restrict__ dec_W, const float* __restrict__ dec_b,
    const float* __restrict__ rew_W, const float* __restrict__ con_W,
    const float* __restrict__ con_b,
    float* __restrict__ xl, float* __restrict__ rl, float* __restrict__ cl,
    float* __restrict__ zpost)
{
  __shared__ float s_stage[8][16 * ZC_];
  const int wave = threadIdx.x >> 5;
  const int lane = threadIdx.x & 31;
  const size_t m0 = ((size_t)blockIdx.x * 8 + wave) * 16;
  float* stage = s_stage[wave];
  const int m     = lane & 15;
  const int kbase = (lane < 16) ? 0 : 16;
  const int rbase = (lane < 16) ? 0 : 8;

  v16h Ah;
  for (int j = 0; j < 16; ++j)
    Ah[j] = (_Float16)hOut[(m0 + m) * H_ + a_kidx(lane, j)];

  for (int s = 0; s < ZS_; ++s) {
    for (int sub = 0; sub < 2; ++sub) {
      int nt  = 2 * s + sub;
      int col = nt * 16 + m;
      const v16h B = *reinterpret_cast<const v16h*>(dynWh + (size_t)col * H_ + kbase);
      v8f acc = {};
      acc = __builtin_amdgcn_wmma_f32_16x16x32_f16(false, Ah, false, B, (short)0, acc, false, false);
      float db = dyn_b[col];
      for (int r = 0; r < 8; ++r)
        stage[(rbase + r) * ZC_ + sub * 16 + m] = acc[r] + db;
    }
    if (lane < 16) {
      size_t row = m0 + (size_t)lane;
      float best = -1e30f; int ci = 0;
      for (int c = 0; c < ZC_; ++c) {
        float v = stage[lane * ZC_ + c] + gumbel4(0xD1B54A3u, (uint32_t)row, (uint32_t)s, (uint32_t)c);
        if (v > best) { best = v; ci = c; }
      }
      float4* zp = reinterpret_cast<float4*>(zpost + row * ZF_ + s * ZC_);
      for (int q = 0; q < 8; ++q) {
        float4 v4 = make_float4(0.f, 0.f, 0.f, 0.f);
        if ((ci >> 2) == q) ((float*)&v4)[ci & 3] = 1.0f;
        zp[q] = v4;
      }
    }
  }

  // decoder (one-hot z => 32-column gather) + reward + continue
  if (lane < 16) {
    size_t row = m0 + (size_t)lane;
    float hv[H_];
    for (int k = 0; k < H_; ++k) hv[k] = hOut[row * H_ + k];
    float racc = 0.0f, cacc = con_b[0];
    for (int k = 0; k < H_; ++k) { racc += hv[k] * rew_W[k]; cacc += hv[k] * con_W[k]; }
    rl[row] = racc;
    cl[row] = cacc;
    for (int o = 0; o < XC_; ++o) {
      const float* wrow = dec_W + (size_t)o * (H_ + ZF_);
      float acc = dec_b[o];
      for (int k = 0; k < H_; ++k) acc += hv[k] * wrow[k];
      for (int s = 0; s < ZS_; ++s)
        acc += wrow[H_ + s * ZC_ + (int)idxWs[row * ZS_ + s]];
      xl[row * XC_ + o] = acc;
    }
  }
}

// ---------------------------------------------------------------------------
// Weight prep: f32 -> f16 packs + Wih transpose (one shot, tiny).
// ---------------------------------------------------------------------------
__global__ void rssm_prep(const float* __restrict__ enc_W,
                          const float* __restrict__ dyn_W,
                          const float* __restrict__ gru_Wih,
                          _Float16* __restrict__ encWh,
                          _Float16* __restrict__ dynWh,
                          float* __restrict__ WihT)
{
  size_t i = (size_t)blockIdx.x * blockDim.x + threadIdx.x;
  const size_t nEnc = (size_t)ZF_ * KENC_;   // 65536
  const size_t nDyn = (size_t)ZF_ * H_;      // 32768
  const size_t nWih = (size_t)GIN_ * 96;     // 98880
  if (i < nEnc) {
    size_t r = i / KENC_, k = i % KENC_;
    float v = (k < (size_t)(H_ + XC_)) ? enc_W[r * (H_ + XC_) + k] : 0.0f;
    encWh[i] = (_Float16)v;
  } else if (i < nEnc + nDyn) {
    size_t j = i - nEnc;
    dynWh[j] = (_Float16)dyn_W[j];
  } else if (i < nEnc + nDyn + nWih) {
    size_t j = i - nEnc - nDyn;
    size_t col = j / 96, g = j % 96;
    WihT[col * 96 + g] = gru_Wih[g * GIN_ + col];
  }
}

extern "C" void kernel_launch(void* const* d_in, const int* in_sizes, int n_in,
                              void* d_out, int out_size, void* d_ws, size_t ws_size,
                              hipStream_t stream) {
  (void)in_sizes; (void)n_in; (void)out_size; (void)ws_size;
  const float* x     = (const float*)d_in[0];
  const float* a     = (const float*)d_in[1];
  const float* h0    = (const float*)d_in[2];
  const float* enc_W = (const float*)d_in[3];
  const float* enc_b = (const float*)d_in[4];
  const float* gWih  = (const float*)d_in[5];
  const float* gWhh  = (const float*)d_in[6];
  const float* gbih  = (const float*)d_in[7];
  const float* gbhh  = (const float*)d_in[8];
  const float* dec_W = (const float*)d_in[9];
  const float* dec_b = (const float*)d_in[10];
  const float* dyn_W = (const float*)d_in[11];
  const float* dyn_b = (const float*)d_in[12];
  const float* rew_W = (const float*)d_in[13];
  const float* con_W = (const float*)d_in[14];
  const float* con_b = (const float*)d_in[15];

  // Workspace layout (256B-aligned slices)
  char* ws = (char*)d_ws;
  float*         hOut  = (float*)ws;                                  // 8 MB
  _Float16*      encWh = (_Float16*)(ws + 8388608);                   // 128 KB
  _Float16*      dynWh = (_Float16*)(ws + 8388608 + 131072);          // 64 KB
  float*         WihT  = (float*)(ws + 8388608 + 131072 + 65536);     // 386 KB
  unsigned char* idxWs = (unsigned char*)(ws + 8388608 + 131072 + 65536 + 395520); // 2 MB

  // Output layout (reference return order, flat f32)
  float* out   = (float*)d_out;
  float* xl    = out;                                   // [T,N,1,10]
  float* rl    = xl + (size_t)T_ * N_ * XC_;            // [T,N,1]
  float* cl    = rl + (size_t)T_ * N_;                  // [T,N,1]
  float* zOut  = cl + (size_t)T_ * N_;                  // [T,N,32,32]
  float* zpost = zOut + (size_t)T_ * N_ * ZF_;          // [T,N,32,32]

  const size_t prepTot = (size_t)ZF_ * KENC_ + (size_t)ZF_ * H_ + (size_t)GIN_ * 96;
  rssm_prep<<<(unsigned)((prepTot + 255) / 256), 256, 0, stream>>>(
      enc_W, dyn_W, gWih, encWh, dynWh, WihT);
  rssm_scan<<<64, 128, 0, stream>>>(
      x, a, h0, enc_b, gWhh, gbih, gbhh, encWh, WihT, hOut, idxWs, zOut);
  rssm_heads<<<512, 256, 0, stream>>>(
      hOut, idxWs, dynWh, dyn_b, dec_W, dec_b, rew_W, con_W, con_b,
      xl, rl, cl, zpost);
}
